// AttentionHead_21912923144865
// MI455X (gfx1250) — matrix operand order repaired and checked by reading, not compiled, not run on hardware
//
#include <hip/hip_runtime.h>

// Problem constants (from reference)
#define EMBED 2048
#define HEADS 128
#define BATCH 4
#define SEQQ  2048

typedef __attribute__((ext_vector_type(16))) __bf16 bf16x16;
typedef __attribute__((ext_vector_type(8)))  __bf16 bf16x8;
typedef __attribute__((ext_vector_type(8)))  float  f32x8;
typedef __attribute__((ext_vector_type(4)))  float  f32x4;

static __device__ __forceinline__ bf16x16 make_frag(bf16x8 lo, bf16x8 hi) {
  bf16x16 f;
#pragma unroll
  for (int i = 0; i < 8; ++i) { f[i] = lo[i]; f[8 + i] = hi[i]; }
  return f;
}

static __device__ __forceinline__ bf16x16 load_bfrag(const __bf16* p) {
  return make_frag(*(const bf16x8*)p, *(const bf16x8*)(p + 16));
}

// ---------------------------------------------------------------------------
// Kernel 0: fp32 -> bf16 weight conversion (W_Q, W_K, W_V; each [H, E]).
// W_V stays in [H, S] row-major layout == V^T: exactly the contiguous layout
// the WMMA B-fragment wants for P @ V.
// ---------------------------------------------------------------------------
__global__ void w_convert_kernel(const float* __restrict__ src,
                                 __bf16* __restrict__ dst, int n) {
  int i = blockIdx.x * blockDim.x + threadIdx.x;
  if (i < n) dst[i] = (__bf16)src[i];
}

// ---------------------------------------------------------------------------
// Kernel 1: Q = X @ Wq^T and K = X @ Wk^T fused (bf16 out, f32 accumulate).
// 256 threads = 8 waves; each wave owns 16 rows of X and computes BOTH Q and
// K (one A-fragment feeds 16 WMMAs; X read exactly once). B fragments and the
// raw X chunk are software-pipelined (cur/next) so VMEM latency overlaps the
// WMMA burst instead of serializing on s_wait_loadcnt 0.
// ---------------------------------------------------------------------------
__global__ __launch_bounds__(256) void qk_proj_kernel(
    const float*  __restrict__ X,     // [B*S, E] fp32
    const __bf16* __restrict__ Wqb,   // [H, E]  bf16
    const __bf16* __restrict__ Wkb,   // [H, E]  bf16
    __bf16* __restrict__ Qbf,         // [B*S, H]
    __bf16* __restrict__ Kbf)         // [B*S, H]
{
  const int lane  = threadIdx.x & 31;
  const int wave  = threadIdx.x >> 5;
  const int rbase = blockIdx.x * 128 + wave * 16;

  const int hl    = lane & 15;
  const int sel8  = (lane >> 4) * 8;    // A-frag k base (ISA layout)
  const int sel16 = (lane >> 4) * 8;    // B-frag k base (table order, matches A)

  const float* xrow = X + (size_t)(rbase + hl) * EMBED + sel8;
  const __bf16* wqrow = Wqb + (size_t)hl * EMBED + sel16;
  const __bf16* wkrow = Wkb + (size_t)hl * EMBED + sel16;

  f32x8 accq[8] = {};
  f32x8 acck[8] = {};

  // Pipeline stage 0: raw X chunk for kc = 0
  f32x4 xr0 = *(const f32x4*)(xrow + 0);
  f32x4 xr1 = *(const f32x4*)(xrow + 4);
  f32x4 xr2 = *(const f32x4*)(xrow + 16);
  f32x4 xr3 = *(const f32x4*)(xrow + 20);

  for (int kc = 0; kc < EMBED; kc += 32) {
    // Convert the already-loaded X chunk to the A fragment
    bf16x16 afrag;
#pragma unroll
    for (int e = 0; e < 4; ++e) {
      afrag[e]      = (__bf16)xr0[e];
      afrag[4 + e]  = (__bf16)xr1[e];
      afrag[8 + e]  = (__bf16)xr2[e];
      afrag[12 + e] = (__bf16)xr3[e];
    }
    // Prefetch next X chunk: overlaps with the 16 WMMAs below
    if (kc + 32 < EMBED) {
      const float* xn = xrow + kc + 32;
      xr0 = *(const f32x4*)(xn + 0);
      xr1 = *(const f32x4*)(xn + 4);
      xr2 = *(const f32x4*)(xn + 16);
      xr3 = *(const f32x4*)(xn + 20);
    }

    // B-fragment double buffer across the 8 column tiles
    bf16x16 bq_cur = load_bfrag(wqrow + kc);
    bf16x16 bk_cur = load_bfrag(wkrow + kc);
#pragma unroll
    for (int t = 0; t < 8; ++t) {
      bf16x16 bq_nxt = bq_cur, bk_nxt = bk_cur;
      if (t < 7) {
        bq_nxt = load_bfrag(wqrow + (size_t)(t + 1) * 16 * EMBED + kc);
        bk_nxt = load_bfrag(wkrow + (size_t)(t + 1) * 16 * EMBED + kc);
      }
      accq[t] = __builtin_amdgcn_wmma_f32_16x16x32_bf16(
          false, afrag, false, bq_cur, (short)0, accq[t], false, false);
      acck[t] = __builtin_amdgcn_wmma_f32_16x16x32_bf16(
          false, afrag, false, bk_cur, (short)0, acck[t], false, false);
      bq_cur = bq_nxt;
      bk_cur = bk_nxt;
    }
  }

  // C/D layout: VGPR r -> row r (lanes 0-15) or r+8 (lanes 16-31), col = lane&15
  const int orow0 = rbase + sel8;
#pragma unroll
  for (int t = 0; t < 8; ++t)
#pragma unroll
    for (int r = 0; r < 8; ++r) {
      const size_t idx = (size_t)(orow0 + r) * HEADS + t * 16 + hl;
      Qbf[idx] = (__bf16)accq[t][r];
      Kbf[idx] = (__bf16)acck[t][r];
    }
}

// ---------------------------------------------------------------------------
// Kernel 2: fused causal flash attention.
// Grid (S/64, B); block = 128 threads = 4 waves; wave owns 16 query rows and
// the full head dim (8 C-fragments). Key tiles of 64, online softmax in f32.
// P round-trips through wave-private LDS (C-layout -> A-layout). K and V
// fragment loads are software-pipelined against the WMMAs.
// ---------------------------------------------------------------------------
__global__ __launch_bounds__(128) void flash_attn_kernel(
    const __bf16* __restrict__ Qbf,   // [B*S, H]
    const __bf16* __restrict__ Kbf,   // [B*S, H]
    const __bf16* __restrict__ Vbf,   // [H, S]  (= W_V bf16; V[k][h]=Vbf[h*S+k])
    float* __restrict__ Outp)         // [B, S, H] fp32
{
  __shared__ __bf16 Plds[4 * 16 * 64];   // 8 KB, one 16x64 tile per wave

  const int lane  = threadIdx.x & 31;
  const int wave  = threadIdx.x >> 5;
  const int qt    = blockIdx.x;               // query tile (64 rows)
  const size_t rowbase = (size_t)blockIdx.y * SEQQ;
  const int qbase = qt * 64 + wave * 16;

  const int hl    = lane & 15;
  const int sel8  = (lane >> 4) * 8;
  const int sel16 = (lane >> 4) * 8;

  // Q fragments for the 4 k-chunks of the 128-wide head dim (kept resident)
  bf16x16 qfrag[4];
  {
    const __bf16* qp = Qbf + (rowbase + qbase + hl) * HEADS;
#pragma unroll
    for (int kc = 0; kc < 4; ++kc) {
      bf16x8 lo = *(const bf16x8*)(qp + kc * 32 + sel8);
      bf16x8 hi = *(const bf16x8*)(qp + kc * 32 + sel8 + 16);
      qfrag[kc] = make_frag(lo, hi);
    }
  }

  f32x8 o[8] = {};
  float m[8], l[8];
#pragma unroll
  for (int r = 0; r < 8; ++r) { m[r] = -3.0e30f; l[r] = 0.0f; }

  __bf16* pw = Plds + wave * (16 * 64);
  const float scale = 0.088388347648318447f;  // 1/sqrt(128)

  const __bf16* kbase_ptr = Kbf + (rowbase + hl) * HEADS + sel16;
  const __bf16* vbase_ptr = Vbf + (size_t)hl * SEQQ + sel16;

  for (int kt = 0; kt <= qt; ++kt) {
    const int kbase = kt * 64;

    if (kt < qt) {  // prefetch next key tile (global_prefetch_b8)
      __builtin_prefetch((const void*)(Kbf + (rowbase + kbase + 64 + hl) * HEADS), 0, 1);
    }

    // ---- S = Q @ K^T for a 16x64 score tile (pipelined K fragments) --------
    f32x8 s[4] = {};
    // fragment address: key row = kbase + nt*16 (+hl), k-chunk = kc*32 (+sel16)
    bf16x16 kf_cur = load_bfrag(kbase_ptr + (size_t)kbase * HEADS);
#pragma unroll
    for (int i = 0; i < 16; ++i) {
      const int nt = i >> 2, kc = i & 3;
      bf16x16 kf_nxt = kf_cur;
      if (i < 15) {
        const int nn = (i + 1) >> 2, nk = (i + 1) & 3;
        kf_nxt = load_bfrag(kbase_ptr + (size_t)(kbase + nn * 16) * HEADS + nk * 32);
      }
      s[nt] = __builtin_amdgcn_wmma_f32_16x16x32_bf16(
          false, qfrag[kc], false, kf_cur, (short)0, s[nt], false, false);
      kf_cur = kf_nxt;
    }

    // ---- scale + causal mask + per-row max ---------------------------------
    float mt[8];
#pragma unroll
    for (int r = 0; r < 8; ++r) mt[r] = -3.0e30f;
#pragma unroll
    for (int nt = 0; nt < 4; ++nt) {
      const int col = kbase + nt * 16 + hl;
#pragma unroll
      for (int r = 0; r < 8; ++r) {
        const int row = qbase + sel8 + r;
        float v = s[nt][r] * scale;
        v = (col > row) ? -3.0e30f : v;   // finite -inf: avoids inf-inf NaN
        s[nt][r] = v;
        mt[r] = fmaxf(mt[r], v);
      }
    }
#pragma unroll
    for (int r = 0; r < 8; ++r)
#pragma unroll
      for (int msk = 1; msk < 16; msk <<= 1)   // reduce within each 16-lane half
        mt[r] = fmaxf(mt[r], __shfl_xor(mt[r], msk, 32));

    // ---- online softmax update + O rescale ---------------------------------
#pragma unroll
    for (int r = 0; r < 8; ++r) {
      const float mnew  = fmaxf(m[r], mt[r]);
      const float alpha = __expf(m[r] - mnew);
      m[r] = mnew;
      float ps = 0.0f;
#pragma unroll
      for (int nt = 0; nt < 4; ++nt) {
        float p = __expf(s[nt][r] - mnew);
        s[nt][r] = p;
        ps += p;
      }
      l[r] = l[r] * alpha + ps;   // lane-partial; reduced once at the end
#pragma unroll
      for (int t = 0; t < 8; ++t) o[t][r] *= alpha;
    }

    // ---- P (C-layout) -> wave-private LDS tile, row-major 16x64 bf16 -------
#pragma unroll
    for (int nt = 0; nt < 4; ++nt) {
      const int col = nt * 16 + hl;
#pragma unroll
      for (int r = 0; r < 8; ++r)
        pw[(sel8 + r) * 64 + col] = (__bf16)s[nt][r];
    }
    // LDS is in-order per wave; compiler inserts s_wait_dscnt for the reloads.

    // ---- O += P @ V (A frags hoisted, pipelined V fragments) ---------------
    bf16x16 pf[2];
#pragma unroll
    for (int kc2 = 0; kc2 < 2; ++kc2) {
      const __bf16* app = pw + hl * 64 + kc2 * 32 + sel8;
      pf[kc2] = make_frag(*(const bf16x8*)app, *(const bf16x8*)(app + 16));
    }
    bf16x16 vf_cur = load_bfrag(vbase_ptr + kbase);
#pragma unroll
    for (int i = 0; i < 16; ++i) {
      const int kc2 = i >> 3, t = i & 7;
      bf16x16 vf_nxt = vf_cur;
      if (i < 15) {
        const int nk = (i + 1) >> 3, ntp = (i + 1) & 7;
        vf_nxt = load_bfrag(vbase_ptr + (size_t)ntp * 16 * SEQQ + kbase + nk * 32);
      }
      o[t] = __builtin_amdgcn_wmma_f32_16x16x32_bf16(
          false, pf[kc2], false, vf_cur, (short)0, o[t], false, false);
      vf_cur = vf_nxt;
    }
  }

  // ---- finalize: reduce l across the 16-lane halves, normalize, store -----
#pragma unroll
  for (int r = 0; r < 8; ++r)
#pragma unroll
    for (int msk = 1; msk < 16; msk <<= 1)
      l[r] += __shfl_xor(l[r], msk, 32);

#pragma unroll
  for (int t = 0; t < 8; ++t)
#pragma unroll
    for (int r = 0; r < 8; ++r) {
      const int row = qbase + sel8 + r;
      Outp[(rowbase + row) * HEADS + t * 16 + hl] = o[t][r] / l[r];
    }
}

// ---------------------------------------------------------------------------
extern "C" void kernel_launch(void* const* d_in, const int* in_sizes, int n_in,
                              void* d_out, int out_size, void* d_ws, size_t ws_size,
                              hipStream_t stream) {
  const float* X  = (const float*)d_in[0];   // [B,S,E]
  const float* Wq = (const float*)d_in[1];   // [H,E]
  const float* Wk = (const float*)d_in[2];   // [H,E]
  const float* Wv = (const float*)d_in[3];   // [H,E] (E==S)

  char* ws = (char*)d_ws;
  const size_t qkBytes = (size_t)BATCH * SEQQ * HEADS * sizeof(__bf16);  // 2 MB
  const size_t wBytes  = (size_t)HEADS * EMBED * sizeof(__bf16);        // 512 KB
  __bf16* Qbf = (__bf16*)(ws);
  __bf16* Kbf = (__bf16*)(ws + qkBytes);
  __bf16* Wqb = (__bf16*)(ws + 2 * qkBytes);
  __bf16* Wkb = (__bf16*)(ws + 2 * qkBytes + wBytes);
  __bf16* Vbf = (__bf16*)(ws + 2 * qkBytes + 2 * wBytes);
  float* Outp = (float*)d_out;

  const int wn = HEADS * EMBED;
  const dim3 cgrid((wn + 255) / 256);
  hipLaunchKernelGGL(w_convert_kernel, cgrid, dim3(256), 0, stream, Wq, Wqb, wn);
  hipLaunchKernelGGL(w_convert_kernel, cgrid, dim3(256), 0, stream, Wk, Wkb, wn);
  hipLaunchKernelGGL(w_convert_kernel, cgrid, dim3(256), 0, stream, Wv, Vbf, wn);

  hipLaunchKernelGGL(qk_proj_kernel, dim3(BATCH * SEQQ / 128), dim3(256), 0, stream,
                     X, Wqb, Wkb, Qbf, Kbf);
  hipLaunchKernelGGL(flash_attn_kernel, dim3(SEQQ / 64, BATCH), dim3(128), 0, stream,
                     Qbf, Kbf, Vbf, Outp);
}
